// GINMolecularPredictor_49546742726915
// MI455X (gfx1250) — matrix operand order repaired and checked by reading, not compiled, not run on hardware
//
#include <hip/hip_runtime.h>
#include <hip/hip_bf16.h>

// ---------------------------------------------------------------------------
// GIN molecular predictor forward pass for gfx1250 (MI455X).
// GEMMs: bf16 WMMA (v_wmma_f32_16x16x32_bf16), fp32 accumulate, fused
// bias+BN(eval)+ReLU+residual epilogue. Block tile 64x64, 8 waves; each wave
// computes two 16x16 tiles (1 A-frag x 2 B-frags). Ping-pong LDS, K x2.
// All element indices fit int32 (largest tensor 102.4M elems) -> 32-bit
// address math. Aggregation/pooling: f32 atomics (atomicMax-as-int valid:
// pooled values >= 0 by construction).
// ---------------------------------------------------------------------------

typedef __attribute__((ext_vector_type(16))) __bf16 bf16x16;
typedef __attribute__((ext_vector_type(2)))  __bf16 bf16x2;
typedef __attribute__((ext_vector_type(8)))  float  f32x8;

#define TM 64
#define TN 64
#define PITCH 17   // dword pitch (34 bf16): odd word stride -> conflict-free

__device__ __forceinline__ unsigned int f2bf(float f) {
  unsigned int u = __float_as_uint(f);
  return (u + 0x7FFFu + ((u >> 16) & 1u)) >> 16;   // round-nearest-even
}
__device__ __forceinline__ unsigned int pack2bf(float lo, float hi) {
#if __has_builtin(__builtin_amdgcn_cvt_pk_bf16_f32)
  return __builtin_bit_cast(unsigned int, __builtin_amdgcn_cvt_pk_bf16_f32(lo, hi));
#else
  return (f2bf(hi) << 16) | f2bf(lo);
#endif
}

// Stage one 64x32 A tile and 32x64 B tile (bf16-packed) into LDS.
template<bool HAS_A2>
__device__ __forceinline__ void stage_tiles(
    unsigned int* __restrict__ sA, unsigned int* __restrict__ sB,
    const float* __restrict__ A, int lda,
    const float* __restrict__ A2, int lda2, float ca,
    const float* __restrict__ B, int ldb,
    int row0, int col0, int k0, int M, int tid)
{
  // A: 64 rows x 8 k-quads = 512 quads, 2 per thread; 16B-aligned float4 loads
#pragma unroll
  for (int i = 0; i < 2; ++i) {
    int pi = tid + i * 256;
    int r  = pi >> 3;
    int kq = (pi & 7) << 2;
    int gr = row0 + r; if (gr >= M) gr = M - 1;
    float4 v = *(const float4*)(A + gr * lda + k0 + kq);
    if (HAS_A2) {
      const float4 w = *(const float4*)(A2 + gr * lda2 + k0 + kq);
      v.x = ca * v.x + w.x;  v.y = ca * v.y + w.y;
      v.z = ca * v.z + w.z;  v.w = ca * v.w + w.w;
    }
    sA[r * PITCH + (kq >> 1)]     = pack2bf(v.x, v.y);
    sA[r * PITCH + (kq >> 1) + 1] = pack2bf(v.z, v.w);
  }
  // B[K,Nc] row-major: k-pairs are ldb apart; store transposed [col][k]
#pragma unroll
  for (int i = 0; i < 4; ++i) {
    int pi = tid + i * 256;
    int c  = pi & 63;
    int kp = (pi >> 6) << 1;
    int bidx = (k0 + kp) * ldb + col0 + c;
    float x0 = B[bidx];
    float x1 = B[bidx + ldb];
    sB[c * PITCH + (kp >> 1)] = pack2bf(x0, x1);
  }
}

__device__ __forceinline__ bf16x16 load_bfrag(const unsigned int* __restrict__ bp, int half) {
  bf16x16 bfrag;
#pragma unroll
  for (int i = 0; i < 8; ++i) {
    bf16x2 pr = __builtin_bit_cast(bf16x2, bp[half * 8 + i]);
    bfrag[2 * i]     = pr[0];
    bfrag[2 * i + 1] = pr[1];
  }
  return bfrag;
}

// Two 16x16x32 WMMA per wave: one A-frag against two adjacent B column tiles.
__device__ __forceinline__ void wmma_step2(
    const unsigned int* __restrict__ sA, const unsigned int* __restrict__ sB,
    f32x8& acc0, f32x8& acc1, int lane, int wm, int wn)
{
  const int half = lane >> 4;
  bf16x16 afrag;
  {
    const unsigned int* ap = sA + ((wm << 4) + (lane & 15)) * PITCH;
#pragma unroll
    for (int p = 0; p < 8; ++p) {
      int kp = half * 16 + p * 2;
      bf16x2 pr = __builtin_bit_cast(bf16x2, ap[kp >> 1]);
      afrag[2 * p]     = pr[0];
      afrag[2 * p + 1] = pr[1];
    }
  }
  const unsigned int* bp0 = sB + ((wn << 5) + (lane & 15)) * PITCH;
  bf16x16 b0 = load_bfrag(bp0, half);
  bf16x16 b1 = load_bfrag(bp0 + 16 * PITCH, half);
  acc0 = __builtin_amdgcn_wmma_f32_16x16x32_bf16(false, afrag, false, b0, (short)0, acc0, false, false);
  acc1 = __builtin_amdgcn_wmma_f32_16x16x32_bf16(false, afrag, false, b1, (short)0, acc1, false, false);
}

// ---------------------------------------------------------------------------
// C[M,Nc] = ReLU( (Aeff[M,K] @ B[K,Nc] + bias) * s + t ) [+ R]
//   Aeff = HAS_A2 ? (1+eps)*A + A2 : A      (GIN (1+eps)*h + agg fusion)
//   s,t folded from BatchNorm(eval) when HAS_BN, else s=1,t=0.
// Requires K % 64 == 0 and Nc % 64 == 0 (true at all call sites).
// ---------------------------------------------------------------------------
template<bool HAS_A2, bool HAS_BN, bool HAS_RES>
__global__ __launch_bounds__(256) void gemm_bf16_wmma_kernel(
    const float* __restrict__ A,  int lda,
    const float* __restrict__ A2, int lda2, const float* __restrict__ eps_ptr,
    const float* __restrict__ B,  int ldb,
    const float* __restrict__ bias,
    const float* __restrict__ bn_gamma, const float* __restrict__ bn_beta,
    const float* __restrict__ bn_mean,  const float* __restrict__ bn_var,
    const float* __restrict__ Rres, int ldr,
    float* __restrict__ C, int ldc,
    int M, int Nc, int K)
{
  __shared__ unsigned int sA[2][TM * PITCH];
  __shared__ unsigned int sB[2][TN * PITCH];

  const int tid  = threadIdx.x;
  const int lane = tid & 31;
  const int wave = tid >> 5;
  const int wm   = wave & 3;     // wave tile row 0..3
  const int wn   = wave >> 2;    // wave tile col-pair 0..1

  const int row0 = blockIdx.x * TM;
  const int col0 = blockIdx.y * TN;

  const float ca = HAS_A2 ? (1.0f + eps_ptr[0]) : 1.0f;

  f32x8 acc0 = {}, acc1 = {};

  for (int k0 = 0; k0 < K; k0 += 64) {
    stage_tiles<HAS_A2>(sA[0], sB[0], A, lda, A2, lda2, ca, B, ldb,
                        row0, col0, k0, M, tid);
    __syncthreads();
    wmma_step2(sA[0], sB[0], acc0, acc1, lane, wm, wn);
    // per-thread order (ds-reads -> wmma) + this barrier make ping-pong safe
    stage_tiles<HAS_A2>(sA[1], sB[1], A, lda, A2, lda2, ca, B, ldb,
                        row0, col0, k0 + 32, M, tid);
    __syncthreads();
    wmma_step2(sA[1], sB[1], acc0, acc1, lane, wm, wn);
  }

  // ---- epilogue: bias + folded BN + ReLU + residual, two column tiles ----
  const int half = lane >> 4;
  const int rowb = row0 + (wm << 4) + half * 8;      // CDNA5 f32 C/D layout
#pragma unroll
  for (int j = 0; j < 2; ++j) {
    const int col = col0 + (wn << 5) + j * 16 + (lane & 15);
    if (col >= Nc) continue;
    float s = 1.0f, t = 0.0f;
    if (HAS_BN) {
      float inv = rsqrtf(bn_var[col] + 1e-5f);
      s = bn_gamma[col] * inv;
      t = bn_beta[col] - bn_mean[col] * s;
    }
    const float bb = bias[col];
    const f32x8& acc = j ? acc1 : acc0;
    int cidx = rowb * ldc + col;                 // int32: tensors < 2^31 elems
    int ridx = HAS_RES ? (rowb * ldr + col) : 0;
#pragma unroll
    for (int r = 0; r < 8; ++r) {
      if (rowb + r < M) {
        float v = (acc[r] + bb) * s + t;
        v = fmaxf(v, 0.0f);
        if (HAS_RES) v += Rres[ridx];
        C[cidx] = v;
      }
      cidx += ldc;
      ridx += ldr;
    }
  }
}

// ---------------------------------------------------------------------------
__global__ void zero_kernel(float* p, size_t n) {
  size_t i = (size_t)blockIdx.x * blockDim.x + threadIdx.x;
  if (i < n) p[i] = 0.0f;
}

// One wave per edge: agg[dst] += h[src]  (128 channels, float4 per lane)
__global__ __launch_bounds__(256) void agg_kernel(
    const float* __restrict__ h, int ldh,
    const int* __restrict__ src, const int* __restrict__ dst,
    float* __restrict__ agg, int E)
{
  int e = (blockIdx.x * 256 + threadIdx.x) >> 5;
  int lane = threadIdx.x & 31;
  if (e >= E) return;
  int s = src[e], d = dst[e];
  const float4 v = *(const float4*)(h + s * ldh + lane * 4);
  float* out = agg + d * 128 + lane * 4;
  atomicAdd(out + 0, v.x);
  atomicAdd(out + 1, v.y);
  atomicAdd(out + 2, v.z);
  atomicAdd(out + 3, v.w);
}

__global__ void count_kernel(const int* __restrict__ batch, float* __restrict__ counts, int N) {
  int i = blockIdx.x * 256 + threadIdx.x;
  if (i < N) atomicAdd(&counts[batch[i]], 1.0f);
}

// 128 threads per node, each handles 4 of the 512 JK channels.
// hjk values are >= 0 (ReLU + nonneg residual), so int atomicMax == float max.
__global__ __launch_bounds__(256) void pool_scatter_kernel(
    const float* __restrict__ hjk, const int* __restrict__ batch,
    float* __restrict__ gsum, float* __restrict__ gmax, int N)
{
  int idx = blockIdx.x * 256 + threadIdx.x;     // < N*128 = 25.6M, fits int
  int node = idx >> 7;
  int q    = idx & 127;
  if (node >= N) return;
  int g = batch[node];
  const float4 v = *(const float4*)(hjk + node * 512 + q * 4);
  float* sp = gsum + g * 512 + q * 4;
  atomicAdd(sp + 0, v.x); atomicAdd(sp + 1, v.y);
  atomicAdd(sp + 2, v.z); atomicAdd(sp + 3, v.w);
  int* mp = (int*)(gmax + g * 512 + q * 4);
  atomicMax(mp + 0, __float_as_int(v.x));
  atomicMax(mp + 1, __float_as_int(v.y));
  atomicMax(mp + 2, __float_as_int(v.z));
  atomicMax(mp + 3, __float_as_int(v.w));
}

__global__ void pool_finalize_kernel(
    const float* __restrict__ gsum, const float* __restrict__ gmax,
    const float* __restrict__ counts, float* __restrict__ gfeat, int G)
{
  int idx = blockIdx.x * 256 + threadIdx.x;     // < G*512 = 4.2M
  int g = idx >> 9;
  int c = idx & 511;
  if (g >= G) return;
  float cnt = counts[g];
  gfeat[g * 1024 + c]       = gsum[idx] / fmaxf(cnt, 1.0f);
  gfeat[g * 1024 + 512 + c] = (cnt > 0.0f) ? gmax[idx] : 0.0f;
}

// Final 64 -> 1 projection: one wave per graph row.
__global__ __launch_bounds__(256) void final_dot_kernel(
    const float* __restrict__ p3, const float* __restrict__ w,
    const float* __restrict__ b, float* __restrict__ out, int G)
{
  int row  = blockIdx.x * 8 + (threadIdx.x >> 5);
  int lane = threadIdx.x & 31;
  if (row >= G) return;
  const float* r = p3 + row * 64;
  float acc = r[lane] * w[lane] + r[32 + lane] * w[32 + lane];
#pragma unroll
  for (int off = 16; off > 0; off >>= 1)
    acc += __shfl_down(acc, off, 32);
  if (lane == 0) out[row] = acc + b[0];
}

// ---------------------------------------------------------------------------
extern "C" void kernel_launch(void* const* d_in, const int* in_sizes, int n_in,
                              void* d_out, int out_size, void* d_ws, size_t ws_size,
                              hipStream_t stream) {
  (void)in_sizes; (void)n_in; (void)out_size; (void)ws_size;
  const int N = 200000, E = 800000, G = 8192, H = 128, L = 4;

  const float* x     = (const float*)d_in[0];
  const int*   ei    = (const int*)d_in[1];     // [2,E]: src row 0, dst row 1
  const int*   batch = (const int*)d_in[2];
  const int*   src = ei;
  const int*   dst = ei + E;

  const float* embed_w = (const float*)d_in[3];
  const float* embed_b = (const float*)d_in[4];
  // per-layer params: eps, lin1.w, lin1.b, bn1.{g,b,m,v}, lin2.w, lin2.b, norm.{g,b,m,v}
  auto LP = [&](int l, int j) -> const float* { return (const float*)d_in[5 + l * 13 + j]; };
  const int PB = 5 + 13 * L;                    // pred params base = 57
  auto PP = [&](int k) -> const float* { return (const float*)d_in[PB + k]; };

  // ---- workspace carve-out ----
  char* ws = (char*)d_ws;
  size_t off = 0;
  auto carve = [&](size_t elems) -> float* {
    float* p = (float*)(ws + off);
    off += elems * sizeof(float);
    return p;
  };
  float* h0     = carve((size_t)N * H);        // embed output
  float* hjk    = carve((size_t)N * 512);      // JK concat, layer l at col l*128
  float* mbuf   = carve((size_t)N * H);        // edge aggregation
  float* t1     = carve((size_t)N * 256);      // lin1 output
  float* gsum   = carve((size_t)G * 512);
  float* gmax   = carve((size_t)G * 512);
  float* counts = carve((size_t)G);
  float* gfeat  = carve((size_t)G * 1024);
  float* p1     = carve((size_t)G * 256);
  float* p2     = carve((size_t)G * 128);
  float* p3     = carve((size_t)G * 64);
  float* outF   = (float*)d_out;

  const dim3 blk(256);
  auto grid_mn = [](int M, int Nc) { return dim3((M + TM - 1) / TM, (Nc + TN - 1) / TN); };

  // 1) embed: h0 = relu(x @ We + be)
  gemm_bf16_wmma_kernel<false, false, false><<<grid_mn(N, H), blk, 0, stream>>>(
      x, 64, nullptr, 0, nullptr, embed_w, H, embed_b,
      nullptr, nullptr, nullptr, nullptr, nullptr, 0, h0, H, N, H, 64);

  // 2) GIN layers
  const float* h_cur = h0; int ldh = H;
  const float* res   = h0; int ldres = H;
  for (int l = 0; l < L; ++l) {
    zero_kernel<<<dim3((unsigned)(((size_t)N * H + 255) / 256)), blk, 0, stream>>>(mbuf, (size_t)N * H);
    agg_kernel<<<dim3((E + 7) / 8), blk, 0, stream>>>(h_cur, ldh, src, dst, mbuf, E);

    // t1 = relu(bn1( ((1+eps)*h + agg) @ W1 + b1 ))
    gemm_bf16_wmma_kernel<true, true, false><<<grid_mn(N, 256), blk, 0, stream>>>(
        h_cur, ldh, mbuf, H, LP(l, 0), LP(l, 1), 256, LP(l, 2),
        LP(l, 3), LP(l, 4), LP(l, 5), LP(l, 6), nullptr, 0, t1, 256, N, 256, H);

    // out_l = relu(bn( t1 @ W2 + b2 )) + res   -> hjk[:, l*128:(l+1)*128]
    float* outl = hjk + l * H;
    gemm_bf16_wmma_kernel<false, true, true><<<grid_mn(N, H), blk, 0, stream>>>(
        t1, 256, nullptr, 0, nullptr, LP(l, 7), H, LP(l, 8),
        LP(l, 9), LP(l, 10), LP(l, 11), LP(l, 12), res, ldres, outl, 512, N, H, 256);

    h_cur = outl; ldh = 512;
    res   = outl; ldres = 512;
  }

  // 3) mean/max pooling over sorted batch segments
  zero_kernel<<<dim3((unsigned)(((size_t)G * 512 + 255) / 256)), blk, 0, stream>>>(gsum, (size_t)G * 512);
  zero_kernel<<<dim3((unsigned)(((size_t)G * 512 + 255) / 256)), blk, 0, stream>>>(gmax, (size_t)G * 512);
  zero_kernel<<<dim3((G + 255) / 256), blk, 0, stream>>>(counts, (size_t)G);
  count_kernel<<<dim3((N + 255) / 256), blk, 0, stream>>>(batch, counts, N);
  pool_scatter_kernel<<<dim3((N * 128 + 255) / 256), blk, 0, stream>>>(hjk, batch, gsum, gmax, N);
  pool_finalize_kernel<<<dim3((G * 512 + 255) / 256), blk, 0, stream>>>(gsum, gmax, counts, gfeat, G);

  // 4) predictor head
  gemm_bf16_wmma_kernel<false, true, false><<<grid_mn(G, 256), blk, 0, stream>>>(
      gfeat, 1024, nullptr, 0, nullptr, PP(0), 256, PP(1),
      PP(2), PP(3), PP(4), PP(5), nullptr, 0, p1, 256, G, 256, 1024);
  gemm_bf16_wmma_kernel<false, true, false><<<grid_mn(G, 128), blk, 0, stream>>>(
      p1, 256, nullptr, 0, nullptr, PP(6), 128, PP(7),
      PP(8), PP(9), PP(10), PP(11), nullptr, 0, p2, 128, G, 128, 256);
  gemm_bf16_wmma_kernel<false, true, false><<<grid_mn(G, 64), blk, 0, stream>>>(
      p2, 128, nullptr, 0, nullptr, PP(12), 64, PP(13),
      PP(14), PP(15), PP(16), PP(17), nullptr, 0, p3, 64, G, 64, 128);
  final_dot_kernel<<<dim3(G / 8), blk, 0, stream>>>(p3, PP(18), PP(19), outF, G);
}
